// GAT_71846212928265
// MI455X (gfx1250) — matrix operand order repaired and checked by reading, not compiled, not run on hardware
//
#include <hip/hip_runtime.h>
#include <hip/hip_fp16.h>

typedef __attribute__((ext_vector_type(16))) _Float16 v16h;
typedef __attribute__((ext_vector_type(8)))  _Float16 v8h;
typedef __attribute__((ext_vector_type(8)))  float    v8f;

#define N_NODES 50000
#define N_EDGES 800000
#define ET      (N_EDGES + N_NODES)   // edges + self loops
#define IN_F    256
#define HC      512                    // H * C
#define CCH     256                    // channels per head
#define NEG_SLOPE 0.2f

// ---------------------------------------------------------------------------
// Order-preserving uint encoding of float for atomicMax-based segment max
// ---------------------------------------------------------------------------
__device__ __forceinline__ unsigned fkey(float f) {
    unsigned b = __float_as_uint(f);
    return (b & 0x80000000u) ? ~b : (b | 0x80000000u);
}
__device__ __forceinline__ float funkey(unsigned k) {
    unsigned b = (k & 0x80000000u) ? (k & 0x7FFFFFFFu) : ~k;
    return __uint_as_float(b);
}

// ---------------------------------------------------------------------------
// wv[k,h] = sum_c W[k, h*256+c] * att[h,c]   (folds attention vec into weights)
// ---------------------------------------------------------------------------
__global__ void gat_wv(const float* __restrict__ W, const float* __restrict__ att,
                       float* __restrict__ wv) {
    int t = blockIdx.x * blockDim.x + threadIdx.x;   // 512 = 256 k * 2 h
    if (t >= IN_F * 2) return;
    int k = t >> 1, h = t & 1;
    const float* wrow = W + (size_t)k * HC + h * CCH;
    const float* arow = att + h * CCH;
    float s = 0.f;
    for (int c = 0; c < CCH; ++c) s += wrow[c] * arow[c];
    wv[k * 2 + h] = s;
}

// ---------------------------------------------------------------------------
// h_src = x @ W_src via v_wmma_f32_16x16x32_f16
// Block tile 128x128, K-step 32, 8 waves; B tile stored in LDS in per-lane
// WMMA fragment order so each fragment is one contiguous 32B ds read.
// ---------------------------------------------------------------------------
#define TM 128
#define TN 128
#define TK 32
#define NTILES (TN / 16)    // 8 B/n-tiles per wave

__global__ __launch_bounds__(256) void gat_gemm_wmma(const float* __restrict__ X,
                                                     const float* __restrict__ W,
                                                     float* __restrict__ Hout) {
    __shared__ _Float16 As[TM][TK + 8];              // row-major [row][K], 80B stride
    __shared__ _Float16 BsF[NTILES * 32 * 16];       // fragment order: [nt][lane][16]

    const int tid  = threadIdx.x;
    const int wave = tid >> 5;
    const int lane = tid & 31;
    const int row0 = blockIdx.y * TM;
    const int col0 = blockIdx.x * TN;

    v8f acc[NTILES];
    #pragma unroll
    for (int nt = 0; nt < NTILES; ++nt) acc[nt] = (v8f){};

    const int arow = wave * 16 + (lane & 15);
    const int akb  = (lane < 16) ? 0 : 16;           // A K-base (contiguous K order, matches B)

    for (int k0 = 0; k0 < IN_F; k0 += TK) {
        // ---- stage A tile (128x32) fp32 -> fp16 : 1024 float4 slots ----
        #pragma unroll
        for (int i = 0; i < 4; ++i) {
            int idx = tid + i * 256;
            int r   = idx >> 3;                      // 8 float4 per row
            int c   = (idx & 7) << 2;
            int gr  = row0 + r;
            float4 v = {0.f, 0.f, 0.f, 0.f};
            if (gr < N_NODES)
                v = *reinterpret_cast<const float4*>(X + (size_t)gr * IN_F + k0 + c);
            As[r][c + 0] = (_Float16)v.x;
            As[r][c + 1] = (_Float16)v.y;
            As[r][c + 2] = (_Float16)v.z;
            As[r][c + 3] = (_Float16)v.w;
        }
        // ---- stage B tile (32x128) fp32 -> fp16, swizzled to fragment order ----
        // element (K=r, col=cc): nt = cc>>4, lane = (cc&15) + (r>=16 ? 16 : 0),
        //                        slot j = r&15  ->  BsF[(nt*32+lane)*16 + j]
        #pragma unroll
        for (int i = 0; i < 4; ++i) {
            int idx = tid + i * 256;                 // 1024 float4 slots
            int r   = idx >> 5;                      // 32 float4 per K-row
            int c   = (idx & 31) << 2;
            float4 v = *reinterpret_cast<const float4*>(W + (size_t)(k0 + r) * HC + col0 + c);
            float vv[4] = {v.x, v.y, v.z, v.w};
            int ladd = (r & 16) ? 16 : 0;
            int j    = r & 15;
            #pragma unroll
            for (int q = 0; q < 4; ++q) {
                int cc = c + q;
                int nt = cc >> 4;
                int lp = (cc & 15) + ladd;
                BsF[((nt * 32 + lp) << 4) + j] = (_Float16)vv[q];
            }
        }
        __syncthreads();

        // ---- A fragment: two aligned 16B chunks -> one v16h ----
        const v8h alo = *reinterpret_cast<const v8h*>(&As[arow][akb]);
        const v8h ahi = *reinterpret_cast<const v8h*>(&As[arow][akb + 8]);
        v16h a = __builtin_shufflevector(alo, ahi,
                 0, 1, 2, 3, 4, 5, 6, 7, 8, 9, 10, 11, 12, 13, 14, 15);

        // ---- B fragments: one contiguous 32B read per tile ----
        const v16h* bf = reinterpret_cast<const v16h*>(BsF);
        #pragma unroll
        for (int nt = 0; nt < NTILES; ++nt) {
            v16h b = bf[nt * 32 + lane];
            acc[nt] = __builtin_amdgcn_wmma_f32_16x16x32_f16(
                false, a, false, b, (short)0, acc[nt], false, false);
        }
        __syncthreads();
    }

    // C/D layout: elem j -> row j (lanes 0-15) or row j+8 (lanes 16-31), col = lane&15
    const int rbase = row0 + wave * 16 + ((lane < 16) ? 0 : 8);
    const int bcl   = lane & 15;
    #pragma unroll
    for (int j = 0; j < 8; ++j) {
        int r = rbase + j;
        if (r < N_NODES) {
            float* o = Hout + (size_t)r * HC + col0 + bcl;
            #pragma unroll
            for (int nt = 0; nt < NTILES; ++nt) o[nt * 16] = acc[nt][j];
        }
    }
}

// ---------------------------------------------------------------------------
// a_src[n,h] = x[n,:] @ wv_src[:,h] ; a_dst likewise (exact fp32, wave/node)
// ---------------------------------------------------------------------------
__global__ __launch_bounds__(256) void gat_attn_coeff(const float* __restrict__ X,
        const float* __restrict__ wvs, const float* __restrict__ wvd,
        float* __restrict__ a_src, float* __restrict__ a_dst) {
    int wave = threadIdx.x >> 5, lane = threadIdx.x & 31;
    int n = blockIdx.x * 8 + wave;
    if (n >= N_NODES) return;
    const float* xr = X + (size_t)n * IN_F;
    float s0 = 0.f, s1 = 0.f, d0 = 0.f, d1 = 0.f;
    for (int k = lane; k < IN_F; k += 32) {
        float xv = xr[k];
        s0 += xv * wvs[k * 2 + 0];
        s1 += xv * wvs[k * 2 + 1];
        d0 += xv * wvd[k * 2 + 0];
        d1 += xv * wvd[k * 2 + 1];
    }
    #pragma unroll
    for (int off = 16; off > 0; off >>= 1) {
        s0 += __shfl_down(s0, off);
        s1 += __shfl_down(s1, off);
        d0 += __shfl_down(d0, off);
        d1 += __shfl_down(d1, off);
    }
    if (lane == 0) {
        a_src[n * 2 + 0] = s0; a_src[n * 2 + 1] = s1;
        a_dst[n * 2 + 0] = d0; a_dst[n * 2 + 1] = d1;
    }
}

// ---------------------------------------------------------------------------
// init kernels
// ---------------------------------------------------------------------------
__global__ void gat_init_nd(unsigned* __restrict__ segmax, float* __restrict__ denom) {
    int i = blockIdx.x * blockDim.x + threadIdx.x;
    if (i < N_NODES * 2) { segmax[i] = 0u; denom[i] = 0.f; }
}
__global__ void gat_init_out(const float* __restrict__ bias, float* __restrict__ out) {
    size_t i = (size_t)blockIdx.x * blockDim.x + threadIdx.x;
    if (i < (size_t)N_NODES * HC) out[i] = bias[i & (HC - 1)];
}

// ---------------------------------------------------------------------------
// edge pass 1: logits + segment max (atomicMax on ordered-uint key)
// ---------------------------------------------------------------------------
__global__ void gat_logits(const int* __restrict__ ei,
                           const float* __restrict__ a_src, const float* __restrict__ a_dst,
                           float* __restrict__ logits, unsigned* __restrict__ segmax) {
    int e = blockIdx.x * blockDim.x + threadIdx.x;
    if (e >= ET) return;
    int s, d;
    if (e < N_EDGES) { s = ei[e]; d = ei[N_EDGES + e]; }
    else             { s = e - N_EDGES; d = s; }
    #pragma unroll
    for (int h = 0; h < 2; ++h) {
        float v = a_src[s * 2 + h] + a_dst[d * 2 + h];
        v = (v > 0.f) ? v : NEG_SLOPE * v;              // leaky_relu
        logits[(size_t)e * 2 + h] = v;
        atomicMax(&segmax[d * 2 + h], fkey(v));
    }
}

// ---------------------------------------------------------------------------
// edge pass 2: denom[d,h] += exp(logit - segmax[d,h])
// ---------------------------------------------------------------------------
__global__ void gat_denom(const int* __restrict__ ei, const float* __restrict__ logits,
                          const unsigned* __restrict__ segmax, float* __restrict__ denom) {
    int e = blockIdx.x * blockDim.x + threadIdx.x;
    if (e >= ET) return;
    int d = (e < N_EDGES) ? ei[N_EDGES + e] : (e - N_EDGES);
    #pragma unroll
    for (int h = 0; h < 2; ++h) {
        float m  = funkey(segmax[d * 2 + h]);
        float ev = __expf(logits[(size_t)e * 2 + h] - m);
        unsafeAtomicAdd(&denom[d * 2 + h], ev);
    }
}

// ---------------------------------------------------------------------------
// edge pass 3: out[d,:] += alpha[e,h] * h_src[s,:]   (block per edge, float4)
// ---------------------------------------------------------------------------
__global__ __launch_bounds__(128) void gat_scatter(const int* __restrict__ ei,
        const float* __restrict__ hsrc, const float* __restrict__ logits,
        const unsigned* __restrict__ segmax, const float* __restrict__ denom,
        float* __restrict__ out) {
    int e = blockIdx.x;
    int s, d;
    if (e < N_EDGES) { s = ei[e]; d = ei[N_EDGES + e]; }
    else             { s = e - N_EDGES; d = s; }
    int c4 = threadIdx.x << 2;            // channel offset 0..508
    int h  = c4 >> 8;                     // head
    float m     = funkey(segmax[d * 2 + h]);
    float dn    = fmaxf(denom[d * 2 + h], 1e-16f);
    float alpha = __expf(logits[(size_t)e * 2 + h] - m) / dn;
    const float4 hv = *reinterpret_cast<const float4*>(hsrc + (size_t)s * HC + c4);
    float* o = out + (size_t)d * HC + c4;
    unsafeAtomicAdd(&o[0], alpha * hv.x);
    unsafeAtomicAdd(&o[1], alpha * hv.y);
    unsafeAtomicAdd(&o[2], alpha * hv.z);
    unsafeAtomicAdd(&o[3], alpha * hv.w);
}

// ---------------------------------------------------------------------------
extern "C" void kernel_launch(void* const* d_in, const int* in_sizes, int n_in,
                              void* d_out, int out_size, void* d_ws, size_t ws_size,
                              hipStream_t stream) {
    const float* x     = (const float*)d_in[0];
    const int*   ei    = (const int*)  d_in[1];   // [2,E] int32
    // d_in[2] edge_attr: unused by reference (edge_dim=None)
    const float* W_src = (const float*)d_in[3];
    const float* W_dst = (const float*)d_in[4];
    const float* att_s = (const float*)d_in[5];
    const float* att_d = (const float*)d_in[6];
    const float* bias  = (const float*)d_in[7];
    float* out = (float*)d_out;

    char* ws = (char*)d_ws;
    size_t off = 0;
    auto alloc = [&](size_t bytes) -> void* {
        void* p = ws + off;
        off = (off + bytes + 255) & ~(size_t)255;
        return p;
    };
    float*    h_src  = (float*)   alloc(sizeof(float) * (size_t)N_NODES * HC); // 102.4 MB
    float*    wv_s   = (float*)   alloc(sizeof(float) * IN_F * 2);
    float*    wv_d   = (float*)   alloc(sizeof(float) * IN_F * 2);
    float*    a_src  = (float*)   alloc(sizeof(float) * N_NODES * 2);
    float*    a_dst  = (float*)   alloc(sizeof(float) * N_NODES * 2);
    unsigned* segmax = (unsigned*)alloc(sizeof(unsigned) * N_NODES * 2);
    float*    denom  = (float*)   alloc(sizeof(float) * N_NODES * 2);
    float*    logits = (float*)   alloc(sizeof(float) * (size_t)ET * 2);       // 6.8 MB

    // fold attention vectors into weights (tiny)
    gat_wv<<<2, 256, 0, stream>>>(W_src, att_s, wv_s);
    gat_wv<<<2, 256, 0, stream>>>(W_dst, att_d, wv_d);

    // main WMMA GEMM: h_src = x @ W_src
    dim3 ggrid(HC / TN, (N_NODES + TM - 1) / TM);   // 4 x 391
    gat_gemm_wmma<<<ggrid, 256, 0, stream>>>(x, W_src, h_src);

    // exact fp32 attention coefficients
    gat_attn_coeff<<<(N_NODES + 7) / 8, 256, 0, stream>>>(x, wv_s, wv_d, a_src, a_dst);

    // init segment state + bias-initialized output
    gat_init_nd<<<(N_NODES * 2 + 255) / 256, 256, 0, stream>>>(segmax, denom);
    gat_init_out<<<(int)(((size_t)N_NODES * HC + 255) / 256), 256, 0, stream>>>(bias, out);

    // segment softmax + aggregation over E + N self-loop edges
    gat_logits <<<(ET + 255) / 256, 256, 0, stream>>>(ei, a_src, a_dst, logits, segmax);
    gat_denom  <<<(ET + 255) / 256, 256, 0, stream>>>(ei, logits, segmax, denom);
    gat_scatter<<<ET, 128, 0, stream>>>(ei, h_src, logits, segmax, denom, out);
}
